// MultiQueryAttention_56358560858478
// MI455X (gfx1250) — matrix operand order repaired and checked by reading, not compiled, not run on hardware
//
#include <hip/hip_runtime.h>
#include <hip/hip_bf16.h>

#define Bdim 2
#define Tdim 2048
#define Ddim 2048
#define Hdim 16
#define HDdim 128

typedef __attribute__((ext_vector_type(16))) __bf16 v16bf;
typedef __attribute__((ext_vector_type(8)))  float  v8f;

union FragBF { v16bf v; unsigned short u[16]; uint4 q[2]; };
union FragF  { v8f v; float f[8]; };

__device__ __forceinline__ unsigned short f2bf(float x) {
  union { float f; unsigned u; } c; c.f = x;
  unsigned r = c.u + 0x7FFFu + ((c.u >> 16) & 1u);   // round-to-nearest-even
  return (unsigned short)(r >> 16);
}

// ---------------------------------------------------------------------------
// fp32 -> bf16 conversion (grid-stride)
// ---------------------------------------------------------------------------
__global__ void f32_to_bf16_kernel(const float* __restrict__ in,
                                   unsigned short* __restrict__ out, int n) {
  int i = blockIdx.x * blockDim.x + threadIdx.x;
  int stride = gridDim.x * blockDim.x;
  for (; i < n; i += stride) out[i] = f2bf(in[i]);
}

// ---------------------------------------------------------------------------
// Generic WMMA GEMM: C[M,N] = A[M,K](bf16) * B[K,N](bf16) + bias
// 128x128 block tile, BK=32, 8 waves (4x2), wave tile 32x64 (2x4 frags).
// B tile stored transposed in LDS so both operands read as contiguous b128.
// ---------------------------------------------------------------------------
template <bool OUT_BF16>
__launch_bounds__(256)
__global__ void gemm_bf16_kernel(const unsigned short* __restrict__ A,
                                 const unsigned short* __restrict__ Bm,
                                 const float* __restrict__ bias,
                                 unsigned short* __restrict__ Cb,
                                 float* __restrict__ Cf,
                                 int M, int N, int K) {
  __shared__ unsigned short As[128 * 48];   // row stride 48 (pad 16) -> 96B, 16B aligned
  __shared__ unsigned short Bs[128 * 48];   // transposed: [n][k]
  const int tid  = threadIdx.x;
  const int lane = tid & 31;
  const int wave = tid >> 5;
  const int lo   = lane & 15;
  const int hi   = lane >> 4;
  const int wm   = wave & 3;
  const int wn   = wave >> 2;
  const int m0   = blockIdx.y * 128;
  const int n0   = blockIdx.x * 128;

  FragF acc[2][4];
  #pragma unroll
  for (int i = 0; i < 2; i++)
    #pragma unroll
    for (int j = 0; j < 4; j++)
      #pragma unroll
      for (int e = 0; e < 8; e++) acc[i][j].f[e] = 0.f;

  const int arow = tid >> 1, ahalf = tid & 1;   // A tile: 128 rows x 32 k
  const int bkk  = tid & 31, bpart = tid >> 5;  // B tile: 32 k x 128 n

  const unsigned short* aptr = A + (long)(m0 + arow) * K + ahalf * 16;
  const unsigned short* bptr = Bm + (long)bkk * N + n0 + bpart * 16;

  for (int k0 = 0; k0 < K; k0 += 32) {
    const uint4* ag = (const uint4*)(aptr + k0);
    uint4 a0 = ag[0], a1 = ag[1];
    const uint4* bg = (const uint4*)(bptr + (long)k0 * N);
    uint4 b0 = bg[0], b1 = bg[1];
    // prefetch next k-tile toward the WGP while this tile computes
    if (k0 + 32 < K) {
      __builtin_prefetch(aptr + k0 + 32, 0, 3);
      __builtin_prefetch(bptr + (long)(k0 + 32) * N, 0, 3);
    }
    __syncthreads();
    *(uint4*)(&As[arow * 48 + ahalf * 16])     = a0;
    *(uint4*)(&As[arow * 48 + ahalf * 16 + 8]) = a1;
    const unsigned short* bu = (const unsigned short*)&b0;
    #pragma unroll
    for (int j = 0; j < 8; j++) Bs[(bpart * 16 + j) * 48 + bkk] = bu[j];
    bu = (const unsigned short*)&b1;
    #pragma unroll
    for (int j = 0; j < 8; j++) Bs[(bpart * 16 + 8 + j) * 48 + bkk] = bu[j];
    __syncthreads();

    FragBF af[2], bfr[4];
    #pragma unroll
    for (int mi = 0; mi < 2; mi++) {
      int r = wm * 32 + mi * 16 + lo;
      af[mi].q[0] = *(const uint4*)(&As[r * 48 + 8 * hi]);       // k = 8h+0..7
      af[mi].q[1] = *(const uint4*)(&As[r * 48 + 16 + 8 * hi]);  // k = 16+8h+0..7
    }
    #pragma unroll
    for (int ni = 0; ni < 4; ni++) {
      int c = wn * 64 + ni * 16 + lo;
      bfr[ni].q[0] = *(const uint4*)(&Bs[c * 48 + 8 * hi]);       // k = 8h+0..7
      bfr[ni].q[1] = *(const uint4*)(&Bs[c * 48 + 16 + 8 * hi]);  // k = 16+8h+0..7
    }
    #pragma unroll
    for (int mi = 0; mi < 2; mi++)
      #pragma unroll
      for (int ni = 0; ni < 4; ni++)
        acc[mi][ni].v = __builtin_amdgcn_wmma_f32_16x16x32_bf16(
            false, af[mi].v, false, bfr[ni].v, (short)0, acc[mi][ni].v, false, false);
  }

  #pragma unroll
  for (int ni = 0; ni < 4; ni++) {
    const int col = n0 + wn * 64 + ni * 16 + lo;
    const float bv = bias[col];
    #pragma unroll
    for (int mi = 0; mi < 2; mi++) {
      const long rbase = (long)(m0 + wm * 32 + mi * 16 + 8 * hi) * N + col;
      #pragma unroll
      for (int r = 0; r < 8; r++) {
        float val = acc[mi][ni].f[r] + bv;
        if (OUT_BF16) Cb[rbase + (long)r * N] = f2bf(val);
        else          Cf[rbase + (long)r * N] = val;
      }
    }
  }
}

// ---------------------------------------------------------------------------
// Causal MQA flash attention. Block = 128 queries of one (b,h); wave = 16 q.
// Computes S' = K*Q^T so the S' C-fragment converts lane-locally into the
// A-fragment of P for P*V (no shuffles / LDS transpose needed).
// ---------------------------------------------------------------------------
__launch_bounds__(256)
__global__ void mqa_attn_kernel(const unsigned short* __restrict__ Q,
                                const unsigned short* __restrict__ Kg,
                                const unsigned short* __restrict__ Vg,
                                unsigned short* __restrict__ O) {
  __shared__ unsigned short Ks[32 * 144];   // [key][d], row stride 144 (288B)
  __shared__ unsigned short Vt[128 * 48];   // [d][key], row stride 48 (96B)
  const int tid  = threadIdx.x;
  const int lane = tid & 31;
  const int wave = tid >> 5;
  const int lo   = lane & 15;
  const int hi   = lane >> 4;
  const int b    = blockIdx.z;
  const int h    = blockIdx.y;
  const int q0b  = blockIdx.x * 128;
  const int q0w  = q0b + wave * 16;
  const int myq  = q0w + lo;                 // this lane's query row
  const float SCALE = 0.08838834764831845f;  // 1/sqrt(128)

  // Preload Q fragments (B operand of K*Q^T): lane = query col, k = d
  FragBF qf[4];
  {
    const unsigned short* qp = Q + (long)(b * Tdim + myq) * Ddim + h * HDdim;
    #pragma unroll
    for (int dc = 0; dc < 4; dc++) {
      qf[dc].q[0] = *(const uint4*)(qp + dc * 32 + 8 * hi);
      qf[dc].q[1] = *(const uint4*)(qp + dc * 32 + 16 + 8 * hi);
    }
  }

  FragF o[8];
  #pragma unroll
  for (int n = 0; n < 8; n++)
    #pragma unroll
    for (int r = 0; r < 8; r++) o[n].f[r] = 0.f;
  float m = -1e30f, l = 0.f;

  const int nchunks = (q0b + 128 + 31) / 32;   // block-uniform causal bound
  const int kkk = tid >> 3, kpart = tid & 7;   // K loader: 32 rows x 8 parts
  const int vkk = tid & 31, vpart = tid >> 5;  // V loader (transpose scatter)

  const unsigned short* kgp = Kg + (long)(b * Tdim + kkk) * HDdim + kpart * 16;
  const unsigned short* vgp = Vg + (long)(b * Tdim + vkk) * HDdim + vpart * 16;

  for (int kc = 0; kc < nchunks; kc++) {
    const int kbase = kc * 32;
    const uint4* kg = (const uint4*)(kgp + (long)kbase * HDdim);
    uint4 k0v = kg[0], k1v = kg[1];
    const uint4* vg = (const uint4*)(vgp + (long)kbase * HDdim);
    uint4 v0v = vg[0], v1v = vg[1];
    if (kc + 1 < nchunks) {   // prefetch next K/V chunk
      __builtin_prefetch(kgp + (long)(kbase + 32) * HDdim, 0, 3);
      __builtin_prefetch(vgp + (long)(kbase + 32) * HDdim, 0, 3);
    }
    __syncthreads();
    *(uint4*)(&Ks[kkk * 144 + kpart * 16])     = k0v;
    *(uint4*)(&Ks[kkk * 144 + kpart * 16 + 8]) = k1v;
    const unsigned short* vu = (const unsigned short*)&v0v;
    #pragma unroll
    for (int j = 0; j < 8; j++) Vt[(vpart * 16 + j) * 48 + vkk] = vu[j];
    vu = (const unsigned short*)&v1v;
    #pragma unroll
    for (int j = 0; j < 8; j++) Vt[(vpart * 16 + 8 + j) * 48 + vkk] = vu[j];
    __syncthreads();

    // S' = K * Q^T: two 16-key tiles; C-frag: lane = query, VGPR row = key.
    // All 8 A-fragments loaded into distinct registers first so the
    // scheduler can stagger s_wait_dscnt across the WMMA chain.
    FragF s0, s1;
    #pragma unroll
    for (int e = 0; e < 8; e++) { s0.f[e] = 0.f; s1.f[e] = 0.f; }
    FragBF ka[2][4];
    #pragma unroll
    for (int dc = 0; dc < 4; dc++) {
      ka[0][dc].q[0] = *(const uint4*)(&Ks[lo * 144 + dc * 32 + 8 * hi]);
      ka[0][dc].q[1] = *(const uint4*)(&Ks[lo * 144 + dc * 32 + 16 + 8 * hi]);
      ka[1][dc].q[0] = *(const uint4*)(&Ks[(16 + lo) * 144 + dc * 32 + 8 * hi]);
      ka[1][dc].q[1] = *(const uint4*)(&Ks[(16 + lo) * 144 + dc * 32 + 16 + 8 * hi]);
    }
    #pragma unroll
    for (int dc = 0; dc < 4; dc++) {
      s0.v = __builtin_amdgcn_wmma_f32_16x16x32_bf16(
          false, ka[0][dc].v, false, qf[dc].v, (short)0, s0.v, false, false);
      s1.v = __builtin_amdgcn_wmma_f32_16x16x32_bf16(
          false, ka[1][dc].v, false, qf[dc].v, (short)0, s1.v, false, false);
    }

    // scale + causal mask + chunk row-max (lane pair L/L^16 holds one query)
    float mloc = -1e30f;
    #pragma unroll
    for (int r = 0; r < 8; r++) {
      int key0 = kbase + r + 8 * hi;
      int key1 = kbase + 16 + r + 8 * hi;
      float a = s0.f[r] * SCALE; if (key0 > myq) a = -1e30f;
      float c = s1.f[r] * SCALE; if (key1 > myq) c = -1e30f;
      s0.f[r] = a; s1.f[r] = c;
      mloc = fmaxf(mloc, fmaxf(a, c));
    }
    mloc = fmaxf(mloc, __shfl_xor(mloc, 16, 32));
    float mnew  = fmaxf(m, mloc);
    float alpha = __expf(m - mnew);
    float psum  = 0.f;
    FragBF pa;   // P as A-fragment: u[0..7] = keys 8h+0..7, u[8..15] = keys 16+8h+0..7
    #pragma unroll
    for (int r = 0; r < 8; r++) {
      float p0 = __expf(s0.f[r] - mnew);
      float p1 = __expf(s1.f[r] - mnew);
      psum += p0 + p1;
      pa.u[r]     = f2bf(p0);
      pa.u[8 + r] = f2bf(p1);
    }
    psum += __shfl_xor(psum, 16, 32);
    l = l * alpha + psum;
    m = mnew;

    // broadcast per-query alpha into O's row dimension (rows live in VGPRs)
    float arow[8];
    #pragma unroll
    for (int r = 0; r < 8; r++) arow[r] = __shfl(alpha, r + 8 * hi, 32);

    // O = O*alpha + P*V   (V B-frag: lane = out dim, k = key, from Vt).
    // Ping-pong vf so WMMA n overlaps the ds_loads of fragment n+1.
    FragBF vf[2];
    vf[0].q[0] = *(const uint4*)(&Vt[lo * 48 + 8 * hi]);
    vf[0].q[1] = *(const uint4*)(&Vt[lo * 48 + 16 + 8 * hi]);
    #pragma unroll
    for (int n = 0; n < 8; n++) {
      if (n < 7) {
        int d = (n + 1) * 16 + lo;
        vf[(n + 1) & 1].q[0] = *(const uint4*)(&Vt[d * 48 + 8 * hi]);
        vf[(n + 1) & 1].q[1] = *(const uint4*)(&Vt[d * 48 + 16 + 8 * hi]);
      }
      #pragma unroll
      for (int r = 0; r < 8; r++) o[n].f[r] *= arow[r];
      o[n].v = __builtin_amdgcn_wmma_f32_16x16x32_bf16(
          false, pa.v, false, vf[n & 1].v, (short)0, o[n].v, false, false);
    }
  }

  // finalize: divide by l, store bf16 into (B,T,D) layout at column h*128
  float linv = 1.f / l;
  float lrow[8];
  #pragma unroll
  for (int r = 0; r < 8; r++) lrow[r] = __shfl(linv, r + 8 * hi, 32);
  unsigned short* op = O + (long)(b * Tdim) * Ddim + h * HDdim;
  #pragma unroll
  for (int n = 0; n < 8; n++)
    #pragma unroll
    for (int r = 0; r < 8; r++) {
      int row = q0w + r + 8 * hi;
      op[(long)row * Ddim + n * 16 + lo] = f2bf(o[n].f[r] * lrow[r]);
    }
}

// ---------------------------------------------------------------------------
extern "C" void kernel_launch(void* const* d_in, const int* in_sizes, int n_in,
                              void* d_out, int out_size, void* d_ws, size_t ws_size,
                              hipStream_t stream) {
  (void)in_sizes; (void)n_in; (void)out_size; (void)ws_size;
  const float* x  = (const float*)d_in[0];
  const float* Wq = (const float*)d_in[1];
  const float* bq = (const float*)d_in[2];
  const float* Wk = (const float*)d_in[3];
  const float* bk = (const float*)d_in[4];
  const float* Wv = (const float*)d_in[5];
  const float* bv = (const float*)d_in[6];
  const float* Wo = (const float*)d_in[7];
  const float* bo = (const float*)d_in[8];
  float* out = (float*)d_out;

  unsigned short* w = (unsigned short*)d_ws;
  const long nx  = (long)Bdim * Tdim * Ddim;    // 8388608
  const long nw  = (long)Ddim * Ddim;           // 4194304
  const long nk  = (long)Ddim * HDdim;          // 262144
  const long nkv = (long)Bdim * Tdim * HDdim;   // 524288
  unsigned short* xb  = w; w += nx;
  unsigned short* wqb = w; w += nw;
  unsigned short* wkb = w; w += nk;
  unsigned short* wvb = w; w += nk;
  unsigned short* wob = w; w += nw;
  unsigned short* qb  = w; w += nx;
  unsigned short* kb  = w; w += nkv;
  unsigned short* vb  = w; w += nkv;
  unsigned short* aob = w; w += nx;   // ~70 MB total workspace

  f32_to_bf16_kernel<<<2048, 256, 0, stream>>>(x,  xb,  (int)nx);
  f32_to_bf16_kernel<<<2048, 256, 0, stream>>>(Wq, wqb, (int)nw);
  f32_to_bf16_kernel<<<512,  256, 0, stream>>>(Wk, wkb, (int)nk);
  f32_to_bf16_kernel<<<512,  256, 0, stream>>>(Wv, wvb, (int)nk);
  f32_to_bf16_kernel<<<2048, 256, 0, stream>>>(Wo, wob, (int)nw);

  const int M = Bdim * Tdim;   // 4096
  // Q = x*Wq + bq  -> bf16 (B,T,D)
  gemm_bf16_kernel<true><<<dim3(Ddim / 128, M / 128), 256, 0, stream>>>(
      xb, wqb, bq, qb, nullptr, M, Ddim, Ddim);
  // K = x*Wk + bk  -> bf16 (B,T,HD)
  gemm_bf16_kernel<true><<<dim3(1, M / 128), 256, 0, stream>>>(
      xb, wkb, bk, kb, nullptr, M, HDdim, Ddim);
  // V = x*Wv + bv  -> bf16 (B,T,HD)
  gemm_bf16_kernel<true><<<dim3(1, M / 128), 256, 0, stream>>>(
      xb, wvb, bv, vb, nullptr, M, HDdim, Ddim);
  // causal MQA attention -> bf16 (B,T,D)
  mqa_attn_kernel<<<dim3(Tdim / 128, Hdim, Bdim), 256, 0, stream>>>(qb, kb, vb, aob);
  // out = attn*Wo + bo -> fp32
  gemm_bf16_kernel<false><<<dim3(Ddim / 128, M / 128), 256, 0, stream>>>(
      aob, wob, bo, nullptr, out, M, Ddim, Ddim);
}
